// LocalTemporalEncoder_80126909874815
// MI455X (gfx1250) — matrix operand order repaired and checked by reading, not compiled, not run on hardware
//
#include <hip/hip_runtime.h>
#include <cstddef>

// Problem constants (match reference)
#define B_   4
#define T_   2048
#define D_   256
#define H_   8
#define DH_  32
#define DFF_ 512
#define MT_  (B_ * T_)   // 8192 total rows (b,t flattened)

typedef __attribute__((ext_vector_type(16))) _Float16 v16h;
typedef __attribute__((ext_vector_type(8)))  _Float16 v8h;
typedef __attribute__((ext_vector_type(8)))  float    v8f;

__device__ __forceinline__ float wave_sum32(float v) {
#pragma unroll
  for (int m = 16; m >= 1; m >>= 1) v += __shfl_xor(v, m, 32);
  return v;
}

// A fragment (16x32 f16, row-major f16 source, K contiguous):
// lane (lo,hi): halves 0..7  = K [kb + hi*16,     +8)
//               halves 8..15 = K [kb + hi*16 + 8, +8)
__device__ __forceinline__ v16h load_a_frag16(const _Float16* __restrict__ row,
                                              int kb, int hi) {
  const v8h c0 = *(const v8h*)(row + kb + hi * 16);       // 16B load
  const v8h c1 = *(const v8h*)(row + kb + hi * 16 + 8);   // 16B load
  v16h a;
#pragma unroll
  for (int i = 0; i < 8; ++i) { a[i] = c0[i]; a[i + 8] = c1[i]; }
  return a;
}

// B fragment (32x16 f16): lane n=lo holds K [kb + hi*16, +16) contiguously:
// one 32-byte load (compiler splits into 2x b128).
__device__ __forceinline__ v16h load_b_frag16(const _Float16* __restrict__ row,
                                              int kb, int hi) {
  return *(const v16h*)(row + kb + hi * 16);
}

// ---------------------------------------------------------------------------
// NT GEMM: out[M,N] = A[M,K] @ W[N,K]^T + bias[N] (+ res[M,N]) (+ ReLU)
// A and W are pre-converted f16; f32 accumulation via v_wmma_f32_16x16x32_f16.
// Each wave owns a 32x32 C macro-tile (2x2 WMMA tiles, 4 independent
// accumulators -> 4 back-to-back WMMAs per K-step). Block = 256 threads =
// 8 waves arranged 4(M) x 2(N): 128M x 64N per block. N,K,flags compile-time.
// OutT selects f32 output (residual/LN consumers) or f16 (GEMM consumers).
// ---------------------------------------------------------------------------
template <int N, int K, bool RES, bool RELU, typename OutT>
__global__ __launch_bounds__(256) void gemm_nt_wmma(
    const _Float16* __restrict__ A, const _Float16* __restrict__ W,
    const float* __restrict__ bias, const float* __restrict__ res,
    OutT* __restrict__ out)
{
  const int lane  = threadIdx.x & 31;
  const int wave  = threadIdx.x >> 5;
  const int lo    = lane & 15;
  const int hi    = lane >> 4;
  const int waveM = wave >> 1;      // 0..3
  const int waveN = wave & 1;       // 0..1

  const int mrow = (blockIdx.y * 4 + waveM) * 32;   // C macro-tile row base
  const int ncol = (blockIdx.x * 2 + waveN) * 32;   // C macro-tile col base

  const _Float16* __restrict__ arow0 = A + (size_t)(mrow + lo) * K;
  const _Float16* __restrict__ arow1 = A + (size_t)(mrow + 16 + lo) * K;
  const _Float16* __restrict__ wrow0 = W + (size_t)(ncol + lo) * K;
  const _Float16* __restrict__ wrow1 = W + (size_t)(ncol + 16 + lo) * K;

  v8f c00 = {0.f,0.f,0.f,0.f,0.f,0.f,0.f,0.f};
  v8f c01 = c00, c10 = c00, c11 = c00;

#pragma unroll
  for (int kb = 0; kb < K; kb += 32) {
    const v16h a0 = load_a_frag16(arow0, kb, hi);
    const v16h a1 = load_a_frag16(arow1, kb, hi);
    const v16h b0 = load_b_frag16(wrow0, kb, hi);
    const v16h b1 = load_b_frag16(wrow1, kb, hi);

    c00 = __builtin_amdgcn_wmma_f32_16x16x32_f16(false, a0, false, b0, (short)0, c00, false, false);
    c01 = __builtin_amdgcn_wmma_f32_16x16x32_f16(false, a0, false, b1, (short)0, c01, false, false);
    c10 = __builtin_amdgcn_wmma_f32_16x16x32_f16(false, a1, false, b0, (short)0, c10, false, false);
    c11 = __builtin_amdgcn_wmma_f32_16x16x32_f16(false, a1, false, b1, (short)0, c11, false, false);
  }

  // C/D layout: VGPR r, lanes 0-15 -> M=r, lanes 16-31 -> M=8+r; N=lane&15
  const float bi0 = bias[ncol + lo];
  const float bi1 = bias[ncol + 16 + lo];

#pragma unroll
  for (int r = 0; r < 8; ++r) {
    const int row0 = mrow + hi * 8 + r;
    const int row1 = row0 + 16;
    const int col0 = ncol + lo;
    const int col1 = col0 + 16;

    float v00 = c00[r] + bi0;
    float v01 = c01[r] + bi1;
    float v10 = c10[r] + bi0;
    float v11 = c11[r] + bi1;
    if (RES) {
      v00 += res[(size_t)row0 * N + col0];
      v01 += res[(size_t)row0 * N + col1];
      v10 += res[(size_t)row1 * N + col0];
      v11 += res[(size_t)row1 * N + col1];
    }
    if (RELU) {
      v00 = fmaxf(v00, 0.f); v01 = fmaxf(v01, 0.f);
      v10 = fmaxf(v10, 0.f); v11 = fmaxf(v11, 0.f);
    }
    out[(size_t)row0 * N + col0] = (OutT)v00;
    out[(size_t)row0 * N + col1] = (OutT)v01;
    out[(size_t)row1 * N + col0] = (OutT)v10;
    out[(size_t)row1 * N + col1] = (OutT)v11;
  }
}

// ---------------------------------------------------------------------------
// Elementwise f32 -> f16 conversion; 8 elements/thread, 16B packed stores.
// n must be a multiple of 2048.
// ---------------------------------------------------------------------------
__global__ __launch_bounds__(256) void cvt_f32_f16(
    const float* __restrict__ src, _Float16* __restrict__ dst)
{
  const size_t idx = ((size_t)blockIdx.x * 256 + threadIdx.x) * 8;
  const float4 a = *(const float4*)(src + idx);
  const float4 b = *(const float4*)(src + idx + 4);
  v8h o;
  o[0] = (_Float16)a.x; o[1] = (_Float16)a.y; o[2] = (_Float16)a.z; o[3] = (_Float16)a.w;
  o[4] = (_Float16)b.x; o[5] = (_Float16)b.y; o[6] = (_Float16)b.z; o[7] = (_Float16)b.w;
  *(v8h*)(dst + idx) = o;
}

// ---------------------------------------------------------------------------
// Windowed attention (WIN=9, clipped-at-edges window == contiguous range
// [max(0,t-4), min(T-1,t+4)]). One wave per (b,t,h); lane = head-dim channel.
// qkv is f16 (produced by GEMM1); all arithmetic fp32; output stored f16
// (consumed only by the out-projection GEMM).
// ---------------------------------------------------------------------------
__global__ __launch_bounds__(256) void win_attn_kernel(
    const _Float16* __restrict__ qkv, _Float16* __restrict__ o)
{
  const int lane  = threadIdx.x & 31;
  const int gwave = (int)((blockIdx.x * 256u + threadIdx.x) >> 5);
  const int h  = gwave & (H_ - 1);
  const int bt = gwave >> 3;          // b*T + t
  const int t  = bt & (T_ - 1);       // T is a power of two

  const _Float16* base = qkv + (size_t)bt * (3 * D_) + h * DH_;
  const float q = (float)base[lane];

  float s[9];
  float mx = -1e30f;
#pragma unroll
  for (int i = 0; i < 9; ++i) {
    const int  d  = i - 4;
    const bool ok = (t + d >= 0) && (t + d < T_);
    const int  dd = ok ? d : 0;
    float p = q * (float)base[(ptrdiff_t)dd * (3 * D_) + D_ + lane];   // k
    p = wave_sum32(p) * 0.17677669529663687f;                          // 1/sqrt(32)
    s[i] = ok ? p : -1e30f;
    mx = fmaxf(mx, s[i]);
  }
  float sum = 0.f;
#pragma unroll
  for (int i = 0; i < 9; ++i) { s[i] = __expf(s[i] - mx); sum += s[i]; }
  const float inv = 1.f / sum;

  float acc = 0.f;
#pragma unroll
  for (int i = 0; i < 9; ++i) {
    const int  d  = i - 4;
    const bool ok = (t + d >= 0) && (t + d < T_);
    const int  dd = ok ? d : 0;
    acc += s[i] * (float)base[(ptrdiff_t)dd * (3 * D_) + 2 * D_ + lane];  // v
  }
  o[(size_t)bt * D_ + h * DH_ + lane] = (_Float16)(acc * inv);
}

// ---------------------------------------------------------------------------
// LayerNorm over D=256; one wave per row, 8 values per lane, butterfly stats.
// DUAL=true additionally writes a packed f16 copy (feeds the next GEMM's A).
// ---------------------------------------------------------------------------
template <bool DUAL>
__global__ __launch_bounds__(256) void layernorm_d256(
    const float* __restrict__ x, const float* __restrict__ g,
    const float* __restrict__ b, float* __restrict__ out,
    _Float16* __restrict__ out16)
{
  const int lane = threadIdx.x & 31;
  const int row  = (int)((blockIdx.x * 256u + threadIdx.x) >> 5);

  const float* xr = x + (size_t)row * D_ + lane * 8;
  const float4 v0 = *(const float4*)xr;
  const float4 v1 = *(const float4*)(xr + 4);
  float vals[8] = {v0.x, v0.y, v0.z, v0.w, v1.x, v1.y, v1.z, v1.w};

  float sm = 0.f, sq = 0.f;
#pragma unroll
  for (int i = 0; i < 8; ++i) { sm += vals[i]; sq += vals[i] * vals[i]; }
  sm = wave_sum32(sm);
  sq = wave_sum32(sq);

  const float mu  = sm * (1.f / D_);
  const float var = sq * (1.f / D_) - mu * mu;   // population variance (jnp.var)
  const float rs  = rsqrtf(var + 1e-5f);

  float* orow = out + (size_t)row * D_ + lane * 8;
  v8h oh;
#pragma unroll
  for (int i = 0; i < 8; ++i) {
    const int col = lane * 8 + i;
    const float v = (vals[i] - mu) * rs * g[col] + b[col];
    orow[i] = v;
    if (DUAL) oh[i] = (_Float16)v;
  }
  if (DUAL) *(v8h*)(out16 + (size_t)row * D_ + lane * 8) = oh;
}

// ---------------------------------------------------------------------------
extern "C" void kernel_launch(void* const* d_in, const int* in_sizes, int n_in,
                              void* d_out, int out_size, void* d_ws, size_t ws_size,
                              hipStream_t stream) {
  (void)in_sizes; (void)n_in; (void)out_size; (void)ws_size;

  const float* features = (const float*)d_in[0];
  const float* w_qkv    = (const float*)d_in[1];
  const float* b_qkv    = (const float*)d_in[2];
  const float* w_out    = (const float*)d_in[3];
  const float* b_out    = (const float*)d_in[4];
  const float* w1       = (const float*)d_in[5];
  const float* b1       = (const float*)d_in[6];
  const float* w2       = (const float*)d_in[7];
  const float* b2       = (const float*)d_in[8];
  const float* g1       = (const float*)d_in[9];
  const float* be1      = (const float*)d_in[10];
  const float* g2       = (const float*)d_in[11];
  const float* be2      = (const float*)d_in[12];
  // d_in[13] = mask: recomputed analytically (clipped window == contiguous range)

  // Workspace (byte offsets; ~43 MB total). qkv16 region is reused for ff16
  // once attention has consumed it; y is reused for y2 once LN1 consumed it.
  char* wsb = (char*)d_ws;
  _Float16* qkv16  = (_Float16*)(wsb + 0);          // [8192, 768] f16 (12.58 MB)
  _Float16* ff16   = (_Float16*)(wsb + 0);          // [8192, 512] f16 (reuse)
  _Float16* o16    = (_Float16*)(wsb + 12582912);   // [8192, 256] f16
  float*    y      = (float*)   (wsb + 16777216);   // [8192, 256] f32
  float*    y2     = y;                             // reuse after LN1
  float*    x1     = (float*)   (wsb + 25165824);   // [8192, 256] f32
  _Float16* x1h    = (_Float16*)(wsb + 33554432);   // [8192, 256] f16
  _Float16* feat16 = (_Float16*)(wsb + 37748736);   // [8192, 256] f16
  _Float16* wqkvh  = (_Float16*)(wsb + 41943040);   // [768, 256] f16
  _Float16* wouth  = (_Float16*)(wsb + 42336256);   // [256, 256] f16
  _Float16* w1h    = (_Float16*)(wsb + 42467328);   // [512, 256] f16
  _Float16* w2h    = (_Float16*)(wsb + 42729472);   // [256, 512] f16
  float*    outp   = (float*)d_out;                 // [8192, 256] f32

  const dim3 blk(256, 1, 1);

  // 0. One-time f32 -> f16 conversions (each element converted exactly once,
  //    instead of once per consuming WMMA tile).
  cvt_f32_f16<<<dim3(2097152 / 2048), blk, 0, stream>>>(features, feat16);
  cvt_f32_f16<<<dim3(196608 / 2048),  blk, 0, stream>>>(w_qkv, wqkvh);
  cvt_f32_f16<<<dim3(65536 / 2048),   blk, 0, stream>>>(w_out, wouth);
  cvt_f32_f16<<<dim3(131072 / 2048),  blk, 0, stream>>>(w1, w1h);
  cvt_f32_f16<<<dim3(131072 / 2048),  blk, 0, stream>>>(w2, w2h);

  // Grid: (N/64, M/128) — each block covers 128 M-rows x 64 N-cols.

  // 1. QKV = X @ Wqkv^T + b_qkv                    [8192,768] -> f16
  gemm_nt_wmma<768, 256, false, false, _Float16>
      <<<dim3(768 / 64, MT_ / 128), blk, 0, stream>>>(
      feat16, wqkvh, b_qkv, nullptr, qkv16);

  // 2. windowed MHSA -> o                           [8192,256] -> f16
  win_attn_kernel<<<dim3((MT_ * H_) / 8), blk, 0, stream>>>(qkv16, o16);

  // 3. y = o @ Wout^T + b_out + features            [8192,256] -> f32
  gemm_nt_wmma<256, 256, true, false, float>
      <<<dim3(256 / 64, MT_ / 128), blk, 0, stream>>>(
      o16, wouth, b_out, features, y);

  // 4. x1 = LN(y; g1, be1) -> f32 + f16 copies
  layernorm_d256<true><<<dim3(MT_ / 8), blk, 0, stream>>>(y, g1, be1, x1, x1h);

  // 5. ff = relu(x1 @ W1^T + b1)                    [8192,512] -> f16
  gemm_nt_wmma<512, 256, false, true, _Float16>
      <<<dim3(512 / 64, MT_ / 128), blk, 0, stream>>>(
      x1h, w1h, b1, nullptr, ff16);

  // 6. y2 = ff @ W2^T + b2 + x1                     [8192,256] -> f32
  gemm_nt_wmma<256, 512, true, false, float>
      <<<dim3(256 / 64, MT_ / 128), blk, 0, stream>>>(
      ff16, w2h, b2, x1, y2);

  // 7. out = LN(y2; g2, be2)
  layernorm_d256<false><<<dim3(MT_ / 8), blk, 0, stream>>>(y2, g2, be2, outp, nullptr);
}
